// MultiHeadAttention_25692494364775
// MI455X (gfx1250) — matrix operand order repaired and checked by reading, not compiled, not run on hardware
//
#include <hip/hip_runtime.h>
#include <hip/hip_bf16.h>

#define BB 2
#define SS 2048
#define DD 1024
#define HH 16
#define DKK 64

typedef __bf16 bf16_t;
typedef __attribute__((ext_vector_type(16))) __bf16 v16bf;
typedef __attribute__((ext_vector_type(8)))  __bf16 v8bf;
typedef __attribute__((ext_vector_type(8)))  float   v8f;
typedef __attribute__((ext_vector_type(4)))  float   v4f;

#define WMMA_BF16(A_, B_, C_) \
  __builtin_amdgcn_wmma_f32_16x16x32_bf16(false, (A_), false, (B_), (short)0, (C_), false, false)

// A-operand 16x32 bf16 fragment from row-major memory (row stride `stride` elems).
// lane L<16 -> row L, elems = K{kofs..+7, kofs+16..+23}; L>=16 -> row L-16, K{+8..+15,+24..+31}.
__device__ __forceinline__ v16bf load_fragA(const bf16_t* __restrict__ base,
                                            int stride, int kofs) {
  const int lane = threadIdx.x & 31;
  const int row  = lane & 15;
  const int kh   = (lane >> 4) << 3;          // 0 or 8
  const bf16_t* p = base + (size_t)row * stride + kofs + kh;
  union { v16bf v; v8bf h[2]; } u;
  u.h[0] = *(const v8bf*)(p);
  u.h[1] = *(const v8bf*)(p + 16);
  return u.v;
}

// B-operand 32x16 bf16 fragment; source stored as 16 rows of N (stride elems), K contiguous.
// lane L<16 -> col L, elems = K{kofs..+7, kofs+16..+23}; L>=16 -> col L-16, K{+8..+15,+24..+31}.
__device__ __forceinline__ v16bf load_fragB(const bf16_t* __restrict__ base,
                                            int stride, int kofs) {
  const int lane = threadIdx.x & 31;
  const int col  = lane & 15;
  const int kh   = (lane >> 4) << 3;          // 0 or 8
  const bf16_t* p = base + (size_t)col * stride + kofs + kh;
  union { v16bf v; v8bf h[2]; } u;
  u.h[0] = *(const v8bf*)(p);
  u.h[1] = *(const v8bf*)(p + 16);
  return u.v;
}

__global__ __launch_bounds__(256) void cvt_f32_bf16(const float* __restrict__ src,
                                                    bf16_t* __restrict__ dst, int n) {
  int i = blockIdx.x * blockDim.x + threadIdx.x;
  if (i < n) dst[i] = (bf16_t)src[i];
}

// Compress int mask [S,S] into bitmask [S, S/32]: bit c of word w = (mask[s, w*32+c] != 0).
__global__ __launch_bounds__(256) void mask_to_bits(const int* __restrict__ mask,
                                                    unsigned* __restrict__ bits) {
  int i = blockIdx.x * blockDim.x + threadIdx.x;   // one word = 32 columns
  if (i >= SS * (SS / 32)) return;
  const int* mp = mask + (size_t)i * 32;
  unsigned bm = 0u;
#pragma unroll
  for (int c = 0; c < 32; ++c) bm |= (mp[c] != 0 ? 1u : 0u) << c;
  bits[i] = bm;
}

// C = (A[M,K] * W[N,K]^T + bias) * oscale  (M=4096, N=1024, K=1024)
// Each wave computes a 32x64 strip: 2 A-frags x 4 B-frags -> 8 WMMAs per 12 loads.
// MODE 0: store bf16 split-head [B,H,S,DK]   (Q, K projections)
// MODE 1: store bf16 transposed [B,H,DK,S]   (V projection)
// MODE 2: store fp32 row-major  [B*S, D]     (output projection)
template<int MODE>
__global__ __launch_bounds__(256, 1) void gemm_bias(const bf16_t* __restrict__ A,
                                                    const bf16_t* __restrict__ W,
                                                    const float* __restrict__ bias,
                                                    void* __restrict__ Out,
                                                    float oscale) {
  const int wave = threadIdx.x >> 5;
  const int lane = threadIdx.x & 31;
  const int tile = blockIdx.x * (blockDim.x >> 5) + wave;
  const int NG   = DD / 64;                    // 16 n-groups of 64
  const int mt   = tile / NG;                  // 0..127 (32-row M tiles)
  const int ng   = tile % NG;                  // 0..15

  const bf16_t* Abase = A + (size_t)mt * 32 * DD;
  const bf16_t* Wbase = W + (size_t)ng * 64 * DD;
  v8f acc[2][4] = {};
  for (int kk = 0; kk < DD; kk += 32) {
    // issue all loads first -> distinct regs -> graduated s_wait_loadcnt
    v16bf a0 = load_fragA(Abase,                    DD, kk);
    v16bf a1 = load_fragA(Abase + (size_t)16 * DD,  DD, kk);
    v16bf b0 = load_fragB(Wbase + (size_t)0  * DD,  DD, kk);
    v16bf b1 = load_fragB(Wbase + (size_t)16 * DD,  DD, kk);
    v16bf b2 = load_fragB(Wbase + (size_t)32 * DD,  DD, kk);
    v16bf b3 = load_fragB(Wbase + (size_t)48 * DD,  DD, kk);
    acc[0][0] = WMMA_BF16(a0, b0, acc[0][0]);
    acc[0][1] = WMMA_BF16(a0, b1, acc[0][1]);
    acc[0][2] = WMMA_BF16(a0, b2, acc[0][2]);
    acc[0][3] = WMMA_BF16(a0, b3, acc[0][3]);
    acc[1][0] = WMMA_BF16(a1, b0, acc[1][0]);
    acc[1][1] = WMMA_BF16(a1, b1, acc[1][1]);
    acc[1][2] = WMMA_BF16(a1, b2, acc[1][2]);
    acc[1][3] = WMMA_BF16(a1, b3, acc[1][3]);
  }

  // C layout: VGPR r, lanes 0-15 -> M=r, N=lane; lanes 16-31 -> M=r+8, N=lane-16
  const int nl   = lane & 15;
  const int mofs = (lane >> 4) << 3;
#pragma unroll
  for (int u = 0; u < 2; ++u) {
#pragma unroll
    for (int t = 0; t < 4; ++t) {
      const int n  = ng * 64 + t * 16 + nl;
      const float bn = bias[n];
#pragma unroll
      for (int r = 0; r < 8; ++r) {
        const int m = mt * 32 + u * 16 + mofs + r;
        const float val = (acc[u][t][r] + bn) * oscale;
        if (MODE == 2) {
          ((float*)Out)[(size_t)m * DD + n] = val;
        } else {
          const int b  = m / SS, s = m % SS;
          const int h  = n >> 6, dk = n & 63;
          size_t off;
          if (MODE == 0) off = (((size_t)(b * HH + h) * SS) + s) * DKK + dk;
          else           off = (((size_t)(b * HH + h) * DKK) + dk) * SS + s;
          ((bf16_t*)Out)[off] = (bf16_t)val;
        }
      }
    }
  }
}

// Flash attention with TRANSPOSED scores (S^T = K * Q^T), log2-domain softmax.
// Q was prescaled by (1/sqrt(DK))*log2(e) in its projection, so scores are already
// in exp2 units: p = exp2(s - m), alpha = exp2(m_old - m_new) -> bare v_exp_f32.
// C element (M=key, N=query): each lane owns 16 of 32 key scores for ONE query ->
// softmax is in-lane + one shfl_xor(16); P tile is already in A-operand layout.
__global__ __launch_bounds__(256, 1) void flash_attn(const bf16_t* __restrict__ Qh,
                                                     const bf16_t* __restrict__ Kh,
                                                     const bf16_t* __restrict__ Vt,
                                                     const unsigned* __restrict__ mbits,
                                                     bf16_t* __restrict__ Ctx) {
  __shared__ __align__(16) float alds[8][16];   // per-wave alpha broadcast (query -> row)

  const int wave  = threadIdx.x >> 5;
  const int lane  = threadIdx.x & 31;
  const int bh    = blockIdx.x;          // 0..B*H-1
  const int b     = bh / HH;
  const int h     = bh % HH;
  const int qbase = blockIdx.y * 128 + wave * 16;
  const int nl    = lane & 15;
  const int mofs  = (lane >> 4) << 3;

  const bf16_t* Qb = Qh + (size_t)bh * SS * DKK + (size_t)qbase * DKK;
  const bf16_t* Kb = Kh + (size_t)bh * SS * DKK;
  const bf16_t* Vb = Vt + (size_t)bh * DKK * SS;
  const unsigned* mrow = mbits + (size_t)(qbase + nl) * (SS / 32);   // this lane's query row

  // Q as B-operand (lane = query), loaded once per d-slice.
  const v16bf qf0 = load_fragB(Qb, DKK, 0);
  const v16bf qf1 = load_fragB(Qb, DKK, 32);

  v8f o[4] = {};
  float m_run = -1e30f, l_run = 0.f;
  float* AL = alds[wave];

  for (int j = 0; j < SS; j += 32) {
    // ---- issue ALL loads for this iteration up front (max MLP) ----
    const unsigned bm = mrow[j >> 5];
    v16bf kf00 = load_fragA(Kb + (size_t)j * DKK,        DKK, 0);
    v16bf kf01 = load_fragA(Kb + (size_t)j * DKK,        DKK, 32);
    v16bf kf10 = load_fragA(Kb + (size_t)(j + 16) * DKK, DKK, 0);
    v16bf kf11 = load_fragA(Kb + (size_t)(j + 16) * DKK, DKK, 32);
    v16bf vf0  = load_fragB(Vb + (size_t)0  * SS, SS, j);
    v16bf vf1  = load_fragB(Vb + (size_t)16 * SS, SS, j);
    v16bf vf2  = load_fragB(Vb + (size_t)32 * SS, SS, j);
    v16bf vf3  = load_fragB(Vb + (size_t)48 * SS, SS, j);

    // s0: keys j..j+15, s1: keys j+16..j+31 (M = key, N = query)
    v8f s0 = {}, s1 = {};
    s0 = WMMA_BF16(kf00, qf0, s0);
    s0 = WMMA_BF16(kf01, qf1, s0);
    s1 = WMMA_BF16(kf10, qf0, s1);
    s1 = WMMA_BF16(kf11, qf1, s1);

    // masking: wave-uniform fast path when the whole 32-key window is unmasked
    if (__any(bm != 0xffffffffu)) {
      const unsigned bq = bm >> mofs;          // bit r -> key mofs+r ; bit 16+r -> key 16+mofs+r
#pragma unroll
      for (int r = 0; r < 8; ++r) {
        if (!((bq >> r) & 1u))        s0[r] = -1e9f;
        if (!((bq >> (16 + r)) & 1u)) s1[r] = -1e9f;
      }
    }

    // online softmax (log2 domain): in-lane reduce + one cross-half shuffle
    float vmax = fmaxf(s0[0], s1[0]);
#pragma unroll
    for (int r = 1; r < 8; ++r) vmax = fmaxf(vmax, fmaxf(s0[r], s1[r]));
    vmax = fmaxf(vmax, __shfl_xor(vmax, 16, 32));
    const float mnew  = fmaxf(m_run, vmax);
    const float alpha = exp2f(m_run - mnew);

    v8bf pk0, pk1;
    float rs = 0.f;
#pragma unroll
    for (int r = 0; r < 8; ++r) {
      const float p0 = exp2f(s0[r] - mnew);
      const float p1 = exp2f(s1[r] - mnew);
      rs += p0 + p1;
      pk0[r] = (bf16_t)p0;
      pk1[r] = (bf16_t)p1;
    }
    rs += __shfl_xor(rs, 16, 32);
    l_run = l_run * alpha + rs;
    m_run = mnew;

    // P fragment (A-operand for P@V) directly in registers:
    //   elems 0..7 = keys mofs..mofs+7, elems 8..15 = keys 16+mofs.. -> exactly pk0|pk1
    union { v16bf v; v8bf h2[2]; } pu;
    pu.h2[0] = pk0;
    pu.h2[1] = pk1;

    // rescale O only when some lane's running max actually moved (alpha==1 is exact otherwise)
    if (__any(alpha != 1.0f)) {
      AL[nl] = alpha;                     // lanes nl and nl+16 write identical value
      asm volatile("s_wait_dscnt 0" ::: "memory");
      const v4f af0 = *(const v4f*)&AL[mofs];
      const v4f af1 = *(const v4f*)&AL[mofs + 4];
#pragma unroll
      for (int t = 0; t < 4; ++t) {
#pragma unroll
        for (int r = 0; r < 4; ++r) { o[t][r] *= af0[r]; o[t][4 + r] *= af1[r]; }
      }
    }

    // O += P @ V
    o[0] = WMMA_BF16(pu.v, vf0, o[0]);
    o[1] = WMMA_BF16(pu.v, vf1, o[1]);
    o[2] = WMMA_BF16(pu.v, vf2, o[2]);
    o[3] = WMMA_BF16(pu.v, vf3, o[3]);
  }

  // normalize: 1/l per query -> per row, broadcast once via LDS
  AL[nl] = 1.0f / l_run;
  asm volatile("s_wait_dscnt 0" ::: "memory");
  const v4f iv0 = *(const v4f*)&AL[mofs];
  const v4f iv1 = *(const v4f*)&AL[mofs + 4];

  // store ctx as [B,S,D] bf16 (feature-contiguous for out-proj)
#pragma unroll
  for (int r = 0; r < 8; ++r) {
    const float inv = (r < 4) ? iv0[r & 3] : iv1[r & 3];
    const int s = qbase + mofs + r;
#pragma unroll
    for (int t = 0; t < 4; ++t) {
      const int d = h * DKK + t * 16 + nl;
      Ctx[((size_t)b * SS + s) * DD + d] = (bf16_t)(o[t][r] * inv);
    }
  }
}

extern "C" void kernel_launch(void* const* d_in, const int* in_sizes, int n_in,
                              void* d_out, int out_size, void* d_ws, size_t ws_size,
                              hipStream_t stream) {
  const float* q    = (const float*)d_in[0];
  const float* k    = (const float*)d_in[1];
  const float* v    = (const float*)d_in[2];
  const int*   mask = (const int*)  d_in[3];
  const float* wq   = (const float*)d_in[4];
  const float* bq   = (const float*)d_in[5];
  const float* wk   = (const float*)d_in[6];
  const float* bk   = (const float*)d_in[7];
  const float* wv   = (const float*)d_in[8];
  const float* bv   = (const float*)d_in[9];
  const float* wo   = (const float*)d_in[10];
  const float* bo   = (const float*)d_in[11];

  const size_t NEin = (size_t)BB * SS * DD;   // 4,194,304
  const size_t NEw  = (size_t)DD * DD;        // 1,048,576
  const size_t NEmb = (size_t)SS * (SS / 32); // 131,072 mask words

  char* w = (char*)d_ws;
  bf16_t* qb  = (bf16_t*)w; w += NEin * 2;
  bf16_t* kb  = (bf16_t*)w; w += NEin * 2;
  bf16_t* vb  = (bf16_t*)w; w += NEin * 2;
  bf16_t* wqb = (bf16_t*)w; w += NEw * 2;
  bf16_t* wkb = (bf16_t*)w; w += NEw * 2;
  bf16_t* wvb = (bf16_t*)w; w += NEw * 2;
  bf16_t* wob = (bf16_t*)w; w += NEw * 2;
  bf16_t* qh  = (bf16_t*)w; w += NEin * 2;
  bf16_t* kh  = (bf16_t*)w; w += NEin * 2;
  bf16_t* vt  = (bf16_t*)w; w += NEin * 2;
  bf16_t* ctx = (bf16_t*)w; w += NEin * 2;
  unsigned* mb = (unsigned*)w; w += NEmb * 4;

  // 1) fp32 -> bf16 staging + mask compression
  {
    const int T = 256;
    cvt_f32_bf16<<<(int)((NEin + T - 1) / T), T, 0, stream>>>(q,  qb,  (int)NEin);
    cvt_f32_bf16<<<(int)((NEin + T - 1) / T), T, 0, stream>>>(k,  kb,  (int)NEin);
    cvt_f32_bf16<<<(int)((NEin + T - 1) / T), T, 0, stream>>>(v,  vb,  (int)NEin);
    cvt_f32_bf16<<<(int)((NEw  + T - 1) / T), T, 0, stream>>>(wq, wqb, (int)NEw);
    cvt_f32_bf16<<<(int)((NEw  + T - 1) / T), T, 0, stream>>>(wk, wkb, (int)NEw);
    cvt_f32_bf16<<<(int)((NEw  + T - 1) / T), T, 0, stream>>>(wv, wvb, (int)NEw);
    cvt_f32_bf16<<<(int)((NEw  + T - 1) / T), T, 0, stream>>>(wo, wob, (int)NEw);
    mask_to_bits<<<(int)((NEmb + T - 1) / T), T, 0, stream>>>(mask, mb);
  }

  // 2) Q/K/V projections. Q carries (1/sqrt(DK)) * log2(e) so flash softmax is exp2-native.
  //    tiles = (4096/32)*(1024/64) = 2048 waves -> 256 blocks of 8 waves.
  {
    const float qscale = 0.125f * 1.4426950408889634f;
    dim3 grid(256), block(256);
    gemm_bias<0><<<grid, block, 0, stream>>>(qb, wqb, bq, (void*)qh, qscale);
    gemm_bias<0><<<grid, block, 0, stream>>>(kb, wkb, bk, (void*)kh, 1.0f);
    gemm_bias<1><<<grid, block, 0, stream>>>(vb, wvb, bv, (void*)vt, 1.0f);
  }

  // 3) fused flash attention (transposed-score, log2-domain softmax)
  {
    dim3 grid(BB * HH, SS / 128), block(256);
    flash_attn<<<grid, block, 0, stream>>>(qh, kh, vt, mb, ctx);
  }

  // 4) output projection (fp32 result)
  {
    dim3 grid(256), block(256);
    gemm_bias<2><<<grid, block, 0, stream>>>(ctx, wob, bo, d_out, 1.0f);
  }
}
